// LocalSlidingWindowAttentionOptimized_25039659336140
// MI455X (gfx1250) — matrix operand order repaired and checked by reading, not compiled, not run on hardware
//
#include <hip/hip_runtime.h>
#include <stdint.h>

// ---------------------------------------------------------------------------
// Sliding-window attention on MI455X (gfx1250), bf16 WMMA 16x16x32 + f32 acc.
// Roofline: ~37 GFLOP vs ~67 MB traffic -> matmul-compute-limited; everything
// fits in the 192 MB L2, so bf16 WMMA throughput is the ceiling that matters.
// ---------------------------------------------------------------------------

typedef __attribute__((ext_vector_type(16))) __bf16 v16bf;
typedef __attribute__((ext_vector_type(8)))  float  v8f;

#define BB 2
#define NN 2048
#define EE 1024
#define HH 16
#define DD 64
#define WWIN 64
#define MROWS (BB * NN) /* 4096 */

union Frag {
  v16bf bf;
  unsigned short u[16];
};

// f32 -> bf16 round-to-nearest-even
__device__ __forceinline__ unsigned short f2bf(float x) {
  unsigned int u = __float_as_uint(x);
  unsigned int r = (u + 0x7FFFu + ((u >> 16) & 1u)) >> 16;
  return (unsigned short)r;
}

// ---------------------------------------------------------------------------
// GEMM:  Y[m][n] = sum_k X[m][k] * Wt[n][k] + bias[n]      (torch Linear)
// block = 256 threads (8 waves), tile = 128 rows x 64 cols, K-step 32.
// MODE 0: f32 out [M, EE]
// MODE 1: bf16 out, permuted to [B][H][N][D]   (for Q, K; scale folded in)
// MODE 2: bf16 out, permuted to [B][H][D][N]   (for V, transposed for P*V)
// Templated so the epilogue/staging have zero runtime branches.
// ---------------------------------------------------------------------------
template <int MODE, bool XBF16>
__global__ __launch_bounds__(256)
void lswa_gemm_wmma(const void* __restrict__ Xv,
                    const float* __restrict__ Wt, const float* __restrict__ bias,
                    void* __restrict__ Out, float out_scale)
{
  // 48-half row stride: 96 bytes, multiple of 16B -> aligned b128 LDS reads.
  __shared__ unsigned short As[128 * 48];
  __shared__ unsigned short Bs[64 * 48];

  const int tid  = threadIdx.x;
  const int lane = tid & 31;
  const int Lo   = lane & 15;   // column / row-in-tile index
  const int g    = lane >> 4;   // half-wave group
  const int wave = tid >> 5;    // 0..7 -> 16-row strip
  const int n0   = blockIdx.x * 64;
  const int m0   = blockIdx.y * 128;

  // accumulators: 4 n-tiles of 16x16 f32 each, init with bias[n]
  v8f c[4];
#pragma unroll
  for (int nt = 0; nt < 4; ++nt) {
    float bv = bias[n0 + nt * 16 + Lo];
#pragma unroll
    for (int r = 0; r < 8; ++r) c[nt][r] = bv;
  }

  for (int k0 = 0; k0 < EE; k0 += 32) {
    // ---- stage A tile (128 x 32) with 128-bit loads --------------------
    if constexpr (!XBF16) {
      const float4* X4 = (const float4*)Xv;
#pragma unroll
      for (int i = 0; i < 4; ++i) {
        int e4 = tid + i * 256;                 // [0, 1024): 128 rows x 8 f4
        int row = e4 >> 3, col4 = e4 & 7;
        float4 v = X4[((size_t)(m0 + row) * EE + k0) / 4 + col4];
        unsigned short* d = &As[row * 48 + col4 * 4];
        d[0] = f2bf(v.x); d[1] = f2bf(v.y); d[2] = f2bf(v.z); d[3] = f2bf(v.w);
      }
    } else {
      const uint4* X4 = (const uint4*)Xv;
#pragma unroll
      for (int i = 0; i < 2; ++i) {
        int e8 = tid + i * 256;                 // [0, 512): 128 rows x 4 u4
        int row = e8 >> 2, col8 = e8 & 3;
        uint4 v = X4[((size_t)(m0 + row) * EE + k0) / 8 + col8];
        *(uint4*)&As[row * 48 + col8 * 8] = v;
      }
    }
    // ---- stage B tile (64 x 32) from weights [EE][EE] ------------------
    {
      const float4* W4 = (const float4*)Wt;
#pragma unroll
      for (int i = 0; i < 2; ++i) {
        int e4 = tid + i * 256;                 // [0, 512): 64 rows x 8 f4
        int row = e4 >> 3, col4 = e4 & 7;
        float4 v = W4[((size_t)(n0 + row) * EE + k0) / 4 + col4];
        unsigned short* d = &Bs[row * 48 + col4 * 4];
        d[0] = f2bf(v.x); d[1] = f2bf(v.y); d[2] = f2bf(v.z); d[3] = f2bf(v.w);
      }
    }

    // ---- prefetch next K-chunk into cache (global_prefetch_b8) ---------
    if (k0 + 32 < EE) {
      int prow = tid >> 1, pcol = (tid & 1) * 16;
      if constexpr (XBF16)
        __builtin_prefetch(&((const unsigned short*)Xv)[(size_t)(m0 + prow) * EE + k0 + 32 + pcol], 0, 0);
      else
        __builtin_prefetch(&((const float*)Xv)[(size_t)(m0 + prow) * EE + k0 + 32 + pcol], 0, 0);
      if (prow < 64)
        __builtin_prefetch(&Wt[(size_t)(n0 + prow) * EE + k0 + 32 + pcol], 0, 0);
    }

    __syncthreads();

    // A fragment (16x32 bf16): lane -> M, halves -> K = 16g + h (contiguous, same as B)
    Frag a;
    const unsigned short* ap = &As[(wave * 16 + Lo) * 48];
#pragma unroll
    for (int hh = 0; hh < 8; ++hh) {
      a.u[hh]     = ap[16 * g + hh];       // K = 16g + hh
      a.u[8 + hh] = ap[16 * g + 8 + hh];   // K = 16g + 8 + hh
    }

#pragma unroll
    for (int nt = 0; nt < 4; ++nt) {
      // B fragment (32x16 bf16): lane -> N, halves -> K = 16g + h
      Frag b;
      const unsigned short* bp = &Bs[(nt * 16 + Lo) * 48];
#pragma unroll
      for (int hh = 0; hh < 16; ++hh) b.u[hh] = bp[16 * g + hh];

      c[nt] = __builtin_amdgcn_wmma_f32_16x16x32_bf16(
          false, a.bf, false, b.bf, (short)0, c[nt], false, false);
    }
    __syncthreads();
  }

  // write-out; C/D layout: VGPR r, group g -> M = r + 8g, N = Lo
#pragma unroll
  for (int nt = 0; nt < 4; ++nt) {
#pragma unroll
    for (int r = 0; r < 8; ++r) {
      int grow = m0 + wave * 16 + r + 8 * g;   // [0, 4096)
      int gcol = n0 + nt * 16 + Lo;            // [0, 1024)
      float v = c[nt][r] * out_scale;
      if constexpr (MODE == 0) {
        ((float*)Out)[(size_t)grow * EE + gcol] = v;
      } else {
        int bb  = grow >> 11;       // / NN
        int tok = grow & (NN - 1);
        int hh  = gcol >> 6;        // / DD
        int dd  = gcol & (DD - 1);
        unsigned short hv = f2bf(v);
        if constexpr (MODE == 1)
          ((unsigned short*)Out)[(((size_t)(bb * HH + hh) * NN) + tok) * DD + dd] = hv;
        else
          ((unsigned short*)Out)[(((size_t)(bb * HH + hh) * DD) + dd) * NN + tok] = hv;
      }
    }
  }
}

// ---------------------------------------------------------------------------
// Banded flash attention: one wave (32 threads) per (b, h, 16-row query block).
// Q pre-scaled by 1/sqrt(D). Key blocks processed in PAIRS so P packs a full
// 16x32 bf16 A-fragment for the P*V WMMA. V is stored transposed [B][H][D][N].
// ---------------------------------------------------------------------------
__global__ __launch_bounds__(32)
void lswa_attn_wmma(const unsigned short* __restrict__ Qb,
                    const unsigned short* __restrict__ Kb,
                    const unsigned short* __restrict__ Vt,
                    unsigned short* __restrict__ Aout)
{
  __shared__ unsigned short Pl[16 * 32];   // wave-private P staging (1 KB)

  const int lane = threadIdx.x & 31;
  const int Lo   = lane & 15;
  const int g    = lane >> 4;

  const int bid  = blockIdx.x;
  const int mblk = bid & 127;          // N/16 = 128
  const int h    = (bid >> 7) & 15;
  const int b    = bid >> 11;
  const int m0   = mblk * 16;

  const unsigned short* qbase = Qb + ((size_t)(b * HH + h) * NN) * DD;
  const unsigned short* kbase = Kb + ((size_t)(b * HH + h) * NN) * DD;
  const unsigned short* vbase = Vt + ((size_t)(b * HH + h) * DD) * NN;

  // Q A-fragments for d = 0..31 and d = 32..63
  Frag aq0, aq1;
  {
    const unsigned short* qr = qbase + (size_t)(m0 + Lo) * DD;
#pragma unroll
    for (int hh = 0; hh < 8; ++hh) {
      aq0.u[hh]     = qr[16 * g + hh];
      aq0.u[8 + hh] = qr[8 + 16 * g + hh];
      aq1.u[hh]     = qr[32 + 16 * g + hh];
      aq1.u[8 + hh] = qr[40 + 16 * g + hh];
    }
  }

  float mrow[8], lrow[8];
  v8f acc[4];
#pragma unroll
  for (int r = 0; r < 8; ++r) { mrow[r] = -__builtin_inff(); lrow[r] = 0.f; }
#pragma unroll
  for (int t = 0; t < 4; ++t)
#pragma unroll
    for (int r = 0; r < 8; ++r) acc[t][r] = 0.f;

  int jlo = m0 - WWIN;      if (jlo < 0) jlo = 0;
  int jhi = m0 + 15 + WWIN; if (jhi > NN - 1) jhi = NN - 1;
  const int kb0 = jlo >> 4, kb1 = jhi >> 4;

  for (int kb = kb0; kb <= kb1; kb += 2) {
    const bool have1 = (kb + 1 <= kb1);   // uniform across the wave
    v8f s0, s1;

    { // scores tile 0: S = Q * K^T  (B[k=d][n=tok] = K[tok][d], contiguous)
      Frag bk0, bk1;
      const unsigned short* kr = kbase + (size_t)(kb * 16 + Lo) * DD;
#pragma unroll
      for (int hh = 0; hh < 16; ++hh) {
        bk0.u[hh] = kr[16 * g + hh];
        bk1.u[hh] = kr[32 + 16 * g + hh];
      }
      v8f z;
#pragma unroll
      for (int r = 0; r < 8; ++r) z[r] = 0.f;
      s0 = __builtin_amdgcn_wmma_f32_16x16x32_bf16(false, aq0.bf, false, bk0.bf, (short)0, z,  false, false);
      s0 = __builtin_amdgcn_wmma_f32_16x16x32_bf16(false, aq1.bf, false, bk1.bf, (short)0, s0, false, false);
    }
    if (have1) { // scores tile 1
      Frag bk0, bk1;
      const unsigned short* kr = kbase + (size_t)((kb + 1) * 16 + Lo) * DD;
#pragma unroll
      for (int hh = 0; hh < 16; ++hh) {
        bk0.u[hh] = kr[16 * g + hh];
        bk1.u[hh] = kr[32 + 16 * g + hh];
      }
      v8f z;
#pragma unroll
      for (int r = 0; r < 8; ++r) z[r] = 0.f;
      s1 = __builtin_amdgcn_wmma_f32_16x16x32_bf16(false, aq0.bf, false, bk0.bf, (short)0, z,  false, false);
      s1 = __builtin_amdgcn_wmma_f32_16x16x32_bf16(false, aq1.bf, false, bk1.bf, (short)0, s1, false, false);
    } else {
#pragma unroll
      for (int r = 0; r < 8; ++r) s1[r] = -__builtin_inff();
    }

    // band mask + online softmax (rows striped: M = r + 8g, N = Lo)
#pragma unroll
    for (int r = 0; r < 8; ++r) {
      int mg = m0 + r + 8 * g;
      int n0g = kb * 16 + Lo;
      int d0 = mg - n0g;        if (d0 < 0) d0 = -d0;
      int d1 = mg - (n0g + 16); if (d1 < 0) d1 = -d1;
      if (d0 > WWIN) s0[r] = -__builtin_inff();
      if (have1 && d1 > WWIN) s1[r] = -__builtin_inff();

      float t = fmaxf(s0[r], s1[r]);
#pragma unroll
      for (int off = 1; off < 16; off <<= 1) t = fmaxf(t, __shfl_xor(t, off, 32));
      float nm   = fmaxf(mrow[r], t);
      float corr = __expf(mrow[r] - nm);
      mrow[r] = nm;
      float p0 = __expf(s0[r] - nm);
      float p1 = __expf(s1[r] - nm);
      float rs = p0 + p1;
#pragma unroll
      for (int off = 1; off < 16; off <<= 1) rs += __shfl_xor(rs, off, 32);
      lrow[r] = lrow[r] * corr + rs;
#pragma unroll
      for (int t2 = 0; t2 < 4; ++t2) acc[t2][r] *= corr;

      Pl[(r + 8 * g) * 32 + Lo]      = f2bf(p0);
      Pl[(r + 8 * g) * 32 + 16 + Lo] = f2bf(p1);
    }

    __syncthreads();   // single-wave WG: S_NOP barrier, forces s_wait_dscnt
    // repack P (16x32) as A-fragment
    Frag ap;
    {
      const unsigned short* pr = &Pl[Lo * 32];
#pragma unroll
      for (int hh = 0; hh < 8; ++hh) {
        ap.u[hh]     = pr[16 * g + hh];
        ap.u[8 + hh] = pr[8 + 16 * g + hh];
      }
    }
    __syncthreads();

    // O += P * V ; V transposed [d][tok] -> contiguous 16-half lane reads
#pragma unroll
    for (int c2 = 0; c2 < 4; ++c2) {
      Frag bv;
      const unsigned short* vr = vbase + (size_t)(c2 * 16 + Lo) * NN + kb * 16;
#pragma unroll
      for (int hh = 0; hh < 16; ++hh) {
        int tok = kb * 16 + 16 * g + hh;
        bv.u[hh] = (tok < NN) ? vr[16 * g + hh] : (unsigned short)0;
      }
      acc[c2] = __builtin_amdgcn_wmma_f32_16x16x32_bf16(
          false, ap.bf, false, bv.bf, (short)0, acc[c2], false, false);
    }
  }

  // normalize and write attention output as bf16 [B][N][E], e = h*64 + d
#pragma unroll
  for (int c2 = 0; c2 < 4; ++c2) {
#pragma unroll
    for (int r = 0; r < 8; ++r) {
      int tok = m0 + r + 8 * g;
      float v = acc[c2][r] / lrow[r];
      Aout[((size_t)(b * NN + tok) * EE) + h * DD + c2 * 16 + Lo] = f2bf(v);
    }
  }
}

// ---------------------------------------------------------------------------
// Second output: float_mask [1,1,N,N] = 0 inside band, -inf outside.
// ---------------------------------------------------------------------------
__global__ __launch_bounds__(256)
void lswa_mask(float* __restrict__ mout)
{
  int idx = blockIdx.x * 256 + threadIdx.x;   // < N*N
  int i = idx >> 11, j = idx & (NN - 1);
  int d = i - j; if (d < 0) d = -d;
  mout[idx] = (d <= WWIN) ? 0.f : -__builtin_inff();
}

// ---------------------------------------------------------------------------
extern "C" void kernel_launch(void* const* d_in, const int* in_sizes, int n_in,
                              void* d_out, int out_size, void* d_ws, size_t ws_size,
                              hipStream_t stream)
{
  (void)in_sizes; (void)n_in; (void)out_size; (void)ws_size;

  const float* q    = (const float*)d_in[0];
  const float* Wq_w = (const float*)d_in[1];
  const float* Wq_b = (const float*)d_in[2];
  const float* Wk_w = (const float*)d_in[3];
  const float* Wk_b = (const float*)d_in[4];
  const float* Wv_w = (const float*)d_in[5];
  const float* Wv_b = (const float*)d_in[6];
  const float* Wo_w = (const float*)d_in[7];
  const float* Wo_b = (const float*)d_in[8];

  const size_t nel = (size_t)BB * HH * NN * DD;     // 4,194,304 elems
  unsigned short* Qb = (unsigned short*)d_ws;       // [B][H][N][D] bf16
  unsigned short* Kb = Qb + nel;                    // [B][H][N][D] bf16
  unsigned short* Vt = Kb + nel;                    // [B][H][D][N] bf16
  unsigned short* At = Vt + nel;                    // [B][N][E]    bf16

  float* out  = (float*)d_out;
  float* mout = out + (size_t)BB * NN * EE;

  const dim3 gg(EE / 64, MROWS / 128);   // (16, 32)
  const dim3 bg(256);
  const float qscale = 0.125f;           // 1/sqrt(D), folded into Q

  lswa_gemm_wmma<1, false><<<gg, bg, 0, stream>>>((const void*)q, Wq_w, Wq_b, (void*)Qb, qscale);
  lswa_gemm_wmma<1, false><<<gg, bg, 0, stream>>>((const void*)q, Wk_w, Wk_b, (void*)Kb, 1.0f);
  lswa_gemm_wmma<2, false><<<gg, bg, 0, stream>>>((const void*)q, Wv_w, Wv_b, (void*)Vt, 1.0f);

  lswa_attn_wmma<<<dim3(BB * HH * (NN / 16)), dim3(32), 0, stream>>>(Qb, Kb, Vt, At);

  lswa_gemm_wmma<0, true><<<gg, bg, 0, stream>>>((const void*)At, Wo_w, Wo_b, (void*)out, 1.0f);

  lswa_mask<<<dim3(NN * NN / 256), dim3(256), 0, stream>>>(mout);
}
